// GAT_20091857011050
// MI455X (gfx1250) — matrix operand order repaired and checked by reading, not compiled, not run on hardware
//
#include <hip/hip_runtime.h>
#include <hip/hip_bf16.h>
#include <math.h>

// ---------------- problem constants (from reference) ----------------
#define NNODES   50000
#define NEDGES   800000
#define ETOT     (NEDGES + NNODES)   // self-loops appended
#define FIN      767
#define HID      256
#define H1       8
#define C1C      32
#define NCLS     10
#define NEG_SLOPE 0.2f

typedef __attribute__((ext_vector_type(16))) _Float16 v16h;
typedef __attribute__((ext_vector_type(8)))  float    v8f;

// ---------------- generic fill ----------------
__global__ __launch_bounds__(256) void fill_kernel(float* __restrict__ p, float v, long n) {
    long i = (long)blockIdx.x * blockDim.x + threadIdx.x;
    if (i < n) p[i] = v;
}

// ---------------- WMMA GEMM: C[M x Nc] = A[M x K] * B[K x Nc] (f32 global, f16 compute, f32 accum)
// 256 threads = 8 waves (4x2); block tile 128x64, K-step 32; wave -> 32x32 (2 A-frags x 2 B-frags,
// 4 WMMAs per K-step). Double-buffered LDS, software pipelined, one barrier per K-step.
#define BM 128
#define BN 64
#define BK 32
__global__ __launch_bounds__(256) void wmma_gemm_kernel(
    const float* __restrict__ A, const float* __restrict__ B, float* __restrict__ C,
    int M, int Nc, int K, int lda, int ldb, int ldc)
{
    __shared__ alignas(16) _Float16 As[2][BM * BK];   // [m][k]          16 KB
    __shared__ alignas(16) _Float16 Bs[2][BN * BK];   // [n][k] (transp)  8 KB

    const int tid  = threadIdx.x;
    const int wid  = tid >> 5;
    const int lane = tid & 31;
    const int block_m = blockIdx.y * BM;
    const int block_n = blockIdx.x * BN;
    const int wm = (wid & 3) * 32;    // wave row offset (0,32,64,96)
    const int wn = (wid >> 2) * 32;   // wave col offset (0,32)

    v8f acc00 = {}, acc01 = {}, acc10 = {}, acc11 = {};

    // staging: A: thread owns 8 contiguous K of rows sa_m and sa_m+64
    const int sa_m   = tid >> 2;              // 0..63
    const int sa_kc  = (tid & 3) * 8;
    const int sa_g0  = block_m + sa_m;
    const int sa_g1  = sa_g0 + 64;
    const bool sa_ok0 = sa_g0 < M;
    const bool sa_ok1 = sa_g1 < M;
    const float* Arow0 = A + (long)(sa_ok0 ? sa_g0 : (M - 1)) * lda;
    const float* Arow1 = A + (long)(sa_ok1 ? sa_g1 : (M - 1)) * lda;
    // B: thread owns 8 contiguous K of one column (lane-coalesced along n per load)
    const int sb_n   = tid & 63;
    const int sb_kc  = (tid >> 6) * 8;
    const int sb_gn  = block_n + sb_n;
    const bool sb_nok = sb_gn < Nc;
    const int sb_gnc = sb_nok ? sb_gn : (Nc - 1);

    auto load_tiles = [&](int k0, float va[16], float vb[8]) {
        #pragma unroll
        for (int j = 0; j < 8; ++j) {
            int gk = k0 + sa_kc + j;
            int gkc = gk < K ? gk : (K - 1);
            va[j]     = Arow0[gkc];
            va[j + 8] = Arow1[gkc];
            if (!(sa_ok0 && gk < K)) va[j]     = 0.0f;
            if (!(sa_ok1 && gk < K)) va[j + 8] = 0.0f;
        }
        #pragma unroll
        for (int j = 0; j < 8; ++j) {
            int gk = k0 + sb_kc + j;
            int gkc = gk < K ? gk : (K - 1);
            vb[j] = B[(long)gkc * ldb + sb_gnc];
            if (!(sb_nok && gk < K)) vb[j] = 0.0f;
        }
    };
    auto store_tiles = [&](int buf, const float va[16], const float vb[8]) {
        union { _Float16 h[8]; uint4 q; } p0, p1, pb;
        #pragma unroll
        for (int j = 0; j < 8; ++j) {
            p0.h[j] = (_Float16)va[j];
            p1.h[j] = (_Float16)va[j + 8];
            pb.h[j] = (_Float16)vb[j];
        }
        *(uint4*)&As[buf][sa_m * BK + sa_kc]        = p0.q;
        *(uint4*)&As[buf][(sa_m + 64) * BK + sa_kc] = p1.q;
        *(uint4*)&Bs[buf][sb_n * BK + sb_kc]        = pb.q;
    };

    // WMMA fragment addressing (ISA 16-bit A 16x32 / B 32x16 lane layouts)
    const int a_row0 = wm + (lane & 15);
    const int a_row1 = a_row0 + 16;
    const int a_k0   = (lane < 16) ? 0 : 8;
    const int a_k1   = a_k0 + 16;
    const int b_n    = lane & 15;
    const int b_k    = (lane < 16) ? 0 : 8;

    const int nIters = (K + BK - 1) / BK;
    float va[16], vb[8];
    load_tiles(0, va, vb);
    store_tiles(0, va, vb);
    __syncthreads();

    for (int it = 0; it < nIters; ++it) {
        const int cur = it & 1;
        const bool hasNext = (it + 1) < nIters;
        if (hasNext) load_tiles((it + 1) * BK, va, vb);   // overlap with WMMAs below

        union { v16h v; uint4 q[2]; } af0, af1, bf0, bf1;
        af0.q[0] = *(const uint4*)&As[cur][a_row0 * BK + a_k0];
        af0.q[1] = *(const uint4*)&As[cur][a_row0 * BK + a_k1];
        af1.q[0] = *(const uint4*)&As[cur][a_row1 * BK + a_k0];
        af1.q[1] = *(const uint4*)&As[cur][a_row1 * BK + a_k1];
        const int n0 = wn + b_n;
        const int n1 = n0 + 16;
        bf0.q[0] = *(const uint4*)&Bs[cur][n0 * BK + b_k];
        bf0.q[1] = *(const uint4*)&Bs[cur][n0 * BK + b_k + 16];
        bf1.q[0] = *(const uint4*)&Bs[cur][n1 * BK + b_k];
        bf1.q[1] = *(const uint4*)&Bs[cur][n1 * BK + b_k + 16];

        acc00 = __builtin_amdgcn_wmma_f32_16x16x32_f16(false, af0.v, false, bf0.v,
                                                       (short)0, acc00, false, false);
        acc01 = __builtin_amdgcn_wmma_f32_16x16x32_f16(false, af0.v, false, bf1.v,
                                                       (short)0, acc01, false, false);
        acc10 = __builtin_amdgcn_wmma_f32_16x16x32_f16(false, af1.v, false, bf0.v,
                                                       (short)0, acc10, false, false);
        acc11 = __builtin_amdgcn_wmma_f32_16x16x32_f16(false, af1.v, false, bf1.v,
                                                       (short)0, acc11, false, false);

        if (hasNext) store_tiles(cur ^ 1, va, vb);        // other buffer: no read conflict
        __syncthreads();                                   // orders 2-deep reuse + next reads
    }

    // epilogue: fast path for interior tiles, guarded path at edges
    const int col0   = block_n + wn + (lane & 15);
    const int rbase0 = block_m + wm + ((lane < 16) ? 0 : 8);   // rows of acc0x
    const int rbase1 = rbase0 + 16;                            // rows of acc1x
    if (block_m + BM <= M && block_n + BN <= Nc) {
        long addr0 = (long)rbase0 * ldc + col0;
        long addr1 = (long)rbase1 * ldc + col0;
        #pragma unroll
        for (int r = 0; r < 8; ++r) {
            C[addr0]      = acc00[r];
            C[addr0 + 16] = acc01[r];
            C[addr1]      = acc10[r];
            C[addr1 + 16] = acc11[r];
            addr0 += ldc; addr1 += ldc;
        }
    } else {
        long addr0 = (long)rbase0 * ldc + col0;
        long addr1 = (long)rbase1 * ldc + col0;
        #pragma unroll
        for (int r = 0; r < 8; ++r) {
            if (rbase0 + r < M) {
                if (col0 < Nc)      C[addr0]      = acc00[r];
                if (col0 + 16 < Nc) C[addr0 + 16] = acc01[r];
            }
            if (rbase1 + r < M) {
                if (col0 < Nc)      C[addr1]      = acc10[r];
                if (col0 + 16 < Nc) C[addr1 + 16] = acc11[r];
            }
            addr0 += ldc; addr1 += ldc;
        }
    }
}

// ---------------- layer-1 attention coefficients: wave per head (wave32) ----------------
__global__ __launch_bounds__(256) void alpha1_kernel(
    const float* __restrict__ h, const float* __restrict__ a_src,
    const float* __restrict__ a_dst, float* __restrict__ as_o, float* __restrict__ ad_o)
{
    const int node = blockIdx.x;
    const int t = threadIdx.x;                 // 8 heads x 32 ch; wave w == head w
    float hv = h[(long)node * HID + t];
    float s = hv * a_src[t];
    float d = hv * a_dst[t];
    #pragma unroll
    for (int off = 16; off > 0; off >>= 1) {
        s += __shfl_down(s, off, 32);
        d += __shfl_down(d, off, 32);
    }
    if ((t & 31) == 0) {
        int head = t >> 5;
        as_o[node * H1 + head] = s;
        ad_o[node * H1 + head] = d;
    }
}

// ---------------- layer-2 attention coefficients (1 head, 10 ch) ----------------
__global__ __launch_bounds__(256) void alpha2_kernel(
    const float* __restrict__ h2, const float* __restrict__ a_src,
    const float* __restrict__ a_dst, float* __restrict__ as_o, float* __restrict__ ad_o)
{
    int i = blockIdx.x * blockDim.x + threadIdx.x;
    if (i >= NNODES) return;
    float s = 0.f, d = 0.f;
    #pragma unroll
    for (int c = 0; c < NCLS; ++c) {
        float v = h2[(long)i * NCLS + c];
        s += v * a_src[c];
        d += v * a_dst[c];
    }
    as_o[i] = s; ad_o[i] = d;
}

__device__ __forceinline__ void edge_ids(int e, int* src, int* dst, const int* __restrict__ ei) {
    if (e < NEDGES) { *src = ei[e]; *dst = ei[NEDGES + e]; }
    else            { *src = e - NEDGES; *dst = e - NEDGES; }
}

__device__ __forceinline__ float leaky(float v) { return v > 0.f ? v : NEG_SLOPE * v; }

// ---------------- segment max over dst (native f32 max atomic) ----------------
template<int H>
__global__ __launch_bounds__(256) void edge_max_kernel(
    const int* __restrict__ ei, const float* __restrict__ as_, const float* __restrict__ ad_,
    float* __restrict__ m)
{
    int e = blockIdx.x * blockDim.x + threadIdx.x;
    if (e >= ETOT) return;
    int src, dst; edge_ids(e, &src, &dst, ei);
    #pragma unroll
    for (int h = 0; h < H; ++h) {
        float v = leaky(as_[src * H + h] + ad_[dst * H + h]);
        __hip_atomic_fetch_max(&m[dst * H + h], v, __ATOMIC_RELAXED, __HIP_MEMORY_SCOPE_AGENT);
    }
}

// ---------------- segment sum of exp(e - m[dst]) ----------------
template<int H>
__global__ __launch_bounds__(256) void edge_sum_kernel(
    const int* __restrict__ ei, const float* __restrict__ as_, const float* __restrict__ ad_,
    const float* __restrict__ m, float* __restrict__ s)
{
    int e = blockIdx.x * blockDim.x + threadIdx.x;
    if (e >= ETOT) return;
    int src, dst; edge_ids(e, &src, &dst, ei);
    #pragma unroll
    for (int h = 0; h < H; ++h) {
        float v = leaky(as_[src * H + h] + ad_[dst * H + h]);
        float w = __expf(v - m[dst * H + h]);
        __hip_atomic_fetch_add(&s[dst * H + h], w, __ATOMIC_RELAXED, __HIP_MEMORY_SCOPE_AGENT);
    }
}

// ---------------- weighted message aggregation, vec4 gather (layer 1; CT divisible by 4)
template<int H, int Cch>
__global__ __launch_bounds__(256) void edge_agg_vec4_kernel(
    const int* __restrict__ ei, const float* __restrict__ as_, const float* __restrict__ ad_,
    const float* __restrict__ m, const float* __restrict__ s,
    const float* __restrict__ hsrc, float* __restrict__ agg, long total4)
{
    long gid = (long)blockIdx.x * blockDim.x + threadIdx.x;
    if (gid >= total4) return;
    constexpr int CT  = H * Cch;     // 256
    constexpr int CT4 = CT / 4;      // 64
    int e  = (int)(gid / CT4);
    int c4 = (int)(gid % CT4) * 4;
    int head = c4 / Cch;             // 4 contiguous channels share one head (Cch % 4 == 0)
    int src, dst; edge_ids(e, &src, &dst, ei);
    float a = leaky(as_[src * H + head] + ad_[dst * H + head]);
    float w = __expf(a - m[dst * H + head]);
    float alpha = w / (s[dst * H + head] + 1e-16f);
    const float4 hv = *(const float4*)&hsrc[(long)src * CT + c4];
    float* dp = &agg[(long)dst * CT + c4];
    __hip_atomic_fetch_add(dp + 0, hv.x * alpha, __ATOMIC_RELAXED, __HIP_MEMORY_SCOPE_AGENT);
    __hip_atomic_fetch_add(dp + 1, hv.y * alpha, __ATOMIC_RELAXED, __HIP_MEMORY_SCOPE_AGENT);
    __hip_atomic_fetch_add(dp + 2, hv.z * alpha, __ATOMIC_RELAXED, __HIP_MEMORY_SCOPE_AGENT);
    __hip_atomic_fetch_add(dp + 3, hv.w * alpha, __ATOMIC_RELAXED, __HIP_MEMORY_SCOPE_AGENT);
}

// ---------------- weighted message aggregation, scalar (layer 2; CT = 10) ----------------
template<int H, int Cch>
__global__ __launch_bounds__(256) void edge_agg_kernel(
    const int* __restrict__ ei, const float* __restrict__ as_, const float* __restrict__ ad_,
    const float* __restrict__ m, const float* __restrict__ s,
    const float* __restrict__ hsrc, float* __restrict__ agg, long total)
{
    long gid = (long)blockIdx.x * blockDim.x + threadIdx.x;
    if (gid >= total) return;
    constexpr int CT = H * Cch;
    int e = (int)(gid / CT);
    int c = (int)(gid % CT);
    int head = c / Cch;
    int src, dst; edge_ids(e, &src, &dst, ei);
    float a = leaky(as_[src * H + head] + ad_[dst * H + head]);
    float w = __expf(a - m[dst * H + head]);
    float alpha = w / (s[dst * H + head] + 1e-16f);
    float val = hsrc[(long)src * CT + c] * alpha;
    __hip_atomic_fetch_add(&agg[(long)dst * CT + c], val, __ATOMIC_RELAXED, __HIP_MEMORY_SCOPE_AGENT);
}

// ---------------- bias + ELU (layer-1 epilogue) ----------------
__global__ __launch_bounds__(256) void bias_elu_kernel(
    const float* __restrict__ agg, const float* __restrict__ b, float* __restrict__ out, long n)
{
    long i = (long)blockIdx.x * blockDim.x + threadIdx.x;
    if (i >= n) return;
    float v = agg[i] + b[i & (HID - 1)];
    out[i] = v > 0.f ? v : (__expf(v) - 1.0f);   // ELU, alpha=1
}

// ---------------- output bias add (in place on d_out) ----------------
__global__ __launch_bounds__(256) void bias_out_kernel(
    float* __restrict__ out, const float* __restrict__ b, long n)
{
    long i = (long)blockIdx.x * blockDim.x + threadIdx.x;
    if (i >= n) return;
    out[i] += b[i % NCLS];
}

static inline int cdiv(long a, long b) { return (int)((a + b - 1) / b); }

extern "C" void kernel_launch(void* const* d_in, const int* in_sizes, int n_in,
                              void* d_out, int out_size, void* d_ws, size_t ws_size,
                              hipStream_t stream) {
    const float* x        = (const float*)d_in[0];
    const int*   ei       = (const int*)  d_in[1];   // [2, E]
    const float* W1       = (const float*)d_in[2];   // [767, 256]
    const float* att_src1 = (const float*)d_in[3];   // [8, 32]
    const float* att_dst1 = (const float*)d_in[4];
    const float* b1       = (const float*)d_in[5];   // [256]
    const float* W2       = (const float*)d_in[6];   // [256, 10]
    const float* att_src2 = (const float*)d_in[7];   // [1, 10]
    const float* att_dst2 = (const float*)d_in[8];
    const float* b2       = (const float*)d_in[9];   // [10]
    float* out = (float*)d_out;                      // [N, 10]

    // workspace carve-up (floats); ~112 MB total
    float* h1   = (float*)d_ws;                  // N*256 (x@W1, later reused for ELU output)
    float* agg1 = h1   + (size_t)NNODES * HID;   // N*256
    float* as1  = agg1 + (size_t)NNODES * HID;   // N*8
    float* ad1  = as1  + (size_t)NNODES * H1;
    float* m1   = ad1  + (size_t)NNODES * H1;
    float* s1   = m1   + (size_t)NNODES * H1;
    float* h2   = s1   + (size_t)NNODES * H1;    // N*10
    float* as2  = h2   + (size_t)NNODES * NCLS;  // N
    float* ad2  = as2  + (size_t)NNODES;
    float* m2   = ad2  + (size_t)NNODES;
    float* s2   = m2   + (size_t)NNODES;

    const long nh  = (long)NNODES * H1;
    const long nhc = (long)NNODES * HID;

    // ---- layer 1 ----
    fill_kernel<<<cdiv(nh, 256),  256, 0, stream>>>(m1, -INFINITY, nh);
    fill_kernel<<<cdiv(nh, 256),  256, 0, stream>>>(s1, 0.0f, nh);
    fill_kernel<<<cdiv(nhc, 256), 256, 0, stream>>>(agg1, 0.0f, nhc);

    dim3 g1(cdiv(HID, BN), cdiv(NNODES, BM));
    wmma_gemm_kernel<<<g1, 256, 0, stream>>>(x, W1, h1, NNODES, HID, FIN, FIN, HID, HID);

    alpha1_kernel<<<NNODES, 256, 0, stream>>>(h1, att_src1, att_dst1, as1, ad1);

    edge_max_kernel<H1><<<cdiv(ETOT, 256), 256, 0, stream>>>(ei, as1, ad1, m1);
    edge_sum_kernel<H1><<<cdiv(ETOT, 256), 256, 0, stream>>>(ei, as1, ad1, m1, s1);
    long tot4 = (long)ETOT * (HID / 4);
    edge_agg_vec4_kernel<H1, C1C><<<cdiv(tot4, 256), 256, 0, stream>>>(ei, as1, ad1, m1, s1,
                                                                       h1, agg1, tot4);
    bias_elu_kernel<<<cdiv(nhc, 256), 256, 0, stream>>>(agg1, b1, h1, nhc);  // h1 := elu(agg1+b1)

    // ---- layer 2 ----
    dim3 g2(cdiv(NCLS, BN), cdiv(NNODES, BM));
    wmma_gemm_kernel<<<g2, 256, 0, stream>>>(h1, W2, h2, NNODES, NCLS, HID, HID, NCLS, NCLS);

    alpha2_kernel<<<cdiv(NNODES, 256), 256, 0, stream>>>(h2, att_src2, att_dst2, as2, ad2);

    fill_kernel<<<cdiv(NNODES, 256), 256, 0, stream>>>(m2, -INFINITY, NNODES);
    fill_kernel<<<cdiv(NNODES, 256), 256, 0, stream>>>(s2, 0.0f, NNODES);
    long no = (long)NNODES * NCLS;
    fill_kernel<<<cdiv(no, 256), 256, 0, stream>>>(out, 0.0f, no);

    edge_max_kernel<1><<<cdiv(ETOT, 256), 256, 0, stream>>>(ei, as2, ad2, m2);
    edge_sum_kernel<1><<<cdiv(ETOT, 256), 256, 0, stream>>>(ei, as2, ad2, m2, s2);
    long tot2 = (long)ETOT * NCLS;
    edge_agg_kernel<1, NCLS><<<cdiv(tot2, 256), 256, 0, stream>>>(ei, as2, ad2, m2, s2,
                                                                  h2, out, tot2);
    bias_out_kernel<<<cdiv(no, 256), 256, 0, stream>>>(out, b2, no);
}
